// RetrievalPolicy_45397804318729
// MI455X (gfx1250) — matrix operand order repaired and checked by reading, not compiled, not run on hardware
//
#include <hip/hip_runtime.h>

// ============================================================================
// RetrievalPolicy GNN (GLASS/GAT) for MI455X (gfx1250, wave32, WMMA).
//
//  * GEMMs: v_wmma_f32_16x16x32_bf16, f32 accumulate. Weights converted +
//    transposed to bf16 once per launch.
//  * A tiles (64x32 f32) staged in LDS via GLOBAL_LOAD_ASYNC_TO_LDS_B128
//    (ASYNCcnt), double-buffered: copy of tile k+1 overlaps WMMA of tile k.
//    This removes the 4x redundant global A reads (4 waves share each M-group).
//  * Dead edge-feature path (edge_input / eq_mix) skipped (-819 MB traffic).
//  * Virtual concats ([x|q], [x|x_in]) via split-K dual A pointers.
//  * trans0/1+mix and comb0/1+mix fused as dual-GEMMs with mix epilogue.
//  * GAT: float atomicMax (int punning) segment-max, atomicAdd segment-sum,
//    f32 atomic scatter (output 51 MB -> L2 resident).
//  * GraphNorm: one-pass column sum/sumsq, affine-derived variance.
//
// INPUT FLATTENING ASSUMPTION (JAX pytree, dict keys sorted):
//   0: edge_input.b 1: edge_input.w (unused)
//   layer l, base = 2 + 23*l:
//     +0 att_dst +1 att_src +2 comb0.b +3 comb0.w +4 comb1.b +5 comb1.w
//     +6 eq_mix.b +7 eq_mix.w (unused) +8 gat_bias +9 gat_lin.b +10 gat_lin.w
//     +11 gn_in.b +12 gn_in.ms +13 gn_in.w +14 gn_out.b +15 gn_out.ms
//     +16 gn_out.w +17 nq_mix.b +18 nq_mix.w +19 trans0.b +20 trans0.w
//     +21 trans1.b +22 trans1.w
//   48: node_input.b 49: node_input.w 50: policy_head.b 51: policy_head.w
//   52: question_input.b 53: question_input.w 54: value1.b 55: value1.w
//   56: value2.b 57: value2.w
//   58: x_  59: edge_index(i32)  60: edge_attr (unused)  61: q_emb  62: mask(u8)
//
// Output: [probs (N)] [state_value (1)] [x (N*H)] [entropy (1)], all f32.
// ============================================================================

#define GN  50000
#define GE  200000
#define GD  1024
#define GH  256
#define GT  (GE + GN)   // edges + self loops
#define ZMIX   0.8f
#define EPSGN  1e-5f
#define SLOPE  0.2f

#define LDS_STRIDE 40                    // padded row stride (floats), bank spread
#define LDS_TILE   (64 * LDS_STRIDE)     // one A tile buffer (floats)

typedef __attribute__((ext_vector_type(16))) __bf16 bf16x16;
typedef __attribute__((ext_vector_type(8)))  float  f32x8;

#define WMMA_BF16(A, B, C) \
  __builtin_amdgcn_wmma_f32_16x16x32_bf16(false, (A), false, (B), (short)0, (C), false, false)

__device__ __forceinline__ __bf16 f2bf(float f) {
  unsigned u = __float_as_uint(f);
  u += 0x7fffu + ((u >> 16) & 1u);           // round-to-nearest-even
  unsigned short hs = (unsigned short)(u >> 16);
  __bf16 r;
  __builtin_memcpy(&r, &hs, 2);
  return r;
}

__device__ __forceinline__ void atomicMaxF(float* addr, float val) {
  if (val >= 0.0f) atomicMax((int*)addr, __float_as_int(val));
  else             atomicMin((unsigned int*)addr, __float_as_uint(val));
}

// ---------------------------------------------------------------------------
// Async-stage one 64x32 f32 A tile into LDS (2 x b128 per thread, ASYNCcnt).
// addr[31:0] of a flat LDS address is the raw LDS byte offset (ISA 10.2).
// Virtual concat: kk < K1 -> A1 (stride lda1) else A2 (lda2==0 => broadcast).
// Out-of-range rows clamp to M-1 (valid memory; results discarded later).
// ---------------------------------------------------------------------------
__device__ __forceinline__ void stage_a_async(
    unsigned ldsBase, const float* __restrict__ A1, int lda1,
    const float* __restrict__ A2, int lda2, int K1,
    int blockM, int M, int kk, int tid)
{
  const float* base;
  int lda, kb = kk;
  if (kk < K1) { base = A1; lda = lda1; }
  else         { base = A2; lda = lda2; kb = kk - K1; }
#pragma unroll
  for (int c = 0; c < 2; ++c) {
    int chunk = tid * 2 + c;             // 0..511 (16B chunks)
    int r  = chunk >> 3;                 // 0..63
    int kc = (chunk & 7) * 4;            // 0..28
    int rowg = blockM + r;
    if (rowg > M - 1) rowg = M - 1;
    const float* g = base + (long long)rowg * lda + kb + kc;
    unsigned loff = ldsBase + (unsigned)((r * LDS_STRIDE + kc) * 4);
    asm volatile("global_load_async_to_lds_b128 %0, %1, off"
                 :: "v"(loff), "v"(g) : "memory");
  }
}

// A-fragment (16x32 bf16, ISA 7.12.2) from staged LDS tile: lane row = l16,
// elems 0..15 = K[half*16..+15].
__device__ __forceinline__ bf16x16 lds_a_frag(const float* __restrict__ As,
                                              int r, int half) {
  const float4* p0 = (const float4*)(As + r * LDS_STRIDE + half * 16);
  float4 x0 = p0[0], x1 = p0[1];
  const float4* p1 = (const float4*)(As + r * LDS_STRIDE + half * 16 + 8);
  float4 y0 = p1[0], y1 = p1[1];
  bf16x16 a;
  a[0]  = f2bf(x0.x); a[1]  = f2bf(x0.y); a[2]  = f2bf(x0.z); a[3]  = f2bf(x0.w);
  a[4]  = f2bf(x1.x); a[5]  = f2bf(x1.y); a[6]  = f2bf(x1.z); a[7]  = f2bf(x1.w);
  a[8]  = f2bf(y0.x); a[9]  = f2bf(y0.y); a[10] = f2bf(y0.z); a[11] = f2bf(y0.w);
  a[12] = f2bf(y1.x); a[13] = f2bf(y1.y); a[14] = f2bf(y1.z); a[15] = f2bf(y1.w);
  return a;
}

// B-fragment: 32x16 bf16; lane col = l16, elems 0..15 = K[half*16..+15].
// Weights pre-transposed: Wt[n*K + k] -> the 16 K-values are contiguous (32B).
__device__ __forceinline__ bf16x16 load_b_frag(const __bf16* __restrict__ Wt,
                                               int K, int col, int kk, int half) {
  return *(const bf16x16*)(Wt + (long long)col * K + kk + half * 16);
}

// ---------------------------------------------------------------------------
// GEMM: C[M x 256] = act(A[M x K] @ W[K x 256] + bias). Block = 64 rows,
// 8 waves (2 M-groups x 4 N-groups), each wave 2x4 16x16 WMMA tiles.
// A double-buffered in LDS via async copies.
// ---------------------------------------------------------------------------
__global__ __launch_bounds__(256, 1)
void k_gemm(const float* __restrict__ A1, int lda1,
            const float* __restrict__ A2, int lda2, int K1,
            const __bf16* __restrict__ Wt, int K,
            const float* __restrict__ bias,
            float* __restrict__ C, int M, int relu)
{
  __shared__ float As[2][LDS_TILE];
  const int tid   = threadIdx.x;
  const int lane  = tid & 31;
  const int wave  = tid >> 5;
  const int waveM = wave & 1;
  const int waveN = wave >> 1;
  const int half  = lane >> 4;
  const int l16   = lane & 15;
  const int blockM = blockIdx.x * 64;
  const unsigned asBase = (unsigned)(unsigned long long)(&As[0][0]);

  f32x8 acc[2][4] = {};

  stage_a_async(asBase, A1, lda1, A2, lda2, K1, blockM, M, 0, tid);
  int cur = 0;
  for (int kk = 0; kk < K; kk += 32) {
    if (kk + 32 < K) {
      __syncthreads();   // next buffer free (prev compute done by all waves)
      stage_a_async(asBase + (unsigned)((cur ^ 1) * LDS_TILE * 4),
                    A1, lda1, A2, lda2, K1, blockM, M, kk + 32, tid);
      asm volatile("s_wait_asynccnt 0x2" ::: "memory");  // current tile landed
    } else {
      asm volatile("s_wait_asynccnt 0x0" ::: "memory");
    }
    __syncthreads();     // all waves' staged data visible

    const float* Ac = &As[cur][0];
    bf16x16 a0 = lds_a_frag(Ac, waveM * 32 + l16, half);
    bf16x16 a1 = lds_a_frag(Ac, waveM * 32 + 16 + l16, half);
    bf16x16 bfrag[4];
#pragma unroll
    for (int nt = 0; nt < 4; ++nt)
      bfrag[nt] = load_b_frag(Wt, K, waveN * 64 + nt * 16 + l16, kk, half);
#pragma unroll
    for (int nt = 0; nt < 4; ++nt) {
      acc[0][nt] = WMMA_BF16(a0, bfrag[nt], acc[0][nt]);
      acc[1][nt] = WMMA_BF16(a1, bfrag[nt], acc[1][nt]);
    }
    cur ^= 1;
  }
#pragma unroll
  for (int mt = 0; mt < 2; ++mt) {
    int rbase = blockM + waveM * 32 + mt * 16 + half * 8;
#pragma unroll
    for (int nt = 0; nt < 4; ++nt) {
      int col = waveN * 64 + nt * 16 + l16;
      float bv = bias[col];
#pragma unroll
      for (int r = 0; r < 8; ++r) {
        int row = rbase + r;
        if (row < M) {
          float v = acc[mt][nt][r] + bv;
          if (relu) v = fmaxf(v, 0.0f);
          C[(long long)row * GH + col] = v;
        }
      }
    }
  }
}

// Dual GEMM + masked Z-mix epilogue:
//   y0 = act(A@W0+b0), y1 = act(A@W1+b1)
//   out = mask ? Z*y1+(1-Z)*y0 : Z*y0+(1-Z)*y1
__global__ __launch_bounds__(256, 1)
void k_gemm_dual_mix(const float* __restrict__ A1, int lda1,
                     const float* __restrict__ A2, int lda2, int K1,
                     const __bf16* __restrict__ Wt0, const __bf16* __restrict__ Wt1,
                     int K,
                     const float* __restrict__ b0, const float* __restrict__ b1,
                     const unsigned char* __restrict__ mask,
                     float* __restrict__ Cout, int M, int relu)
{
  __shared__ float As[2][LDS_TILE];
  const int tid   = threadIdx.x;
  const int lane  = tid & 31;
  const int wave  = tid >> 5;
  const int waveM = wave & 1;
  const int waveN = wave >> 1;
  const int half  = lane >> 4;
  const int l16   = lane & 15;
  const int blockM = blockIdx.x * 64;
  const unsigned asBase = (unsigned)(unsigned long long)(&As[0][0]);

  f32x8 acc0[2][4] = {};
  f32x8 acc1[2][4] = {};

  stage_a_async(asBase, A1, lda1, A2, lda2, K1, blockM, M, 0, tid);
  int cur = 0;
  for (int kk = 0; kk < K; kk += 32) {
    if (kk + 32 < K) {
      __syncthreads();
      stage_a_async(asBase + (unsigned)((cur ^ 1) * LDS_TILE * 4),
                    A1, lda1, A2, lda2, K1, blockM, M, kk + 32, tid);
      asm volatile("s_wait_asynccnt 0x2" ::: "memory");
    } else {
      asm volatile("s_wait_asynccnt 0x0" ::: "memory");
    }
    __syncthreads();

    const float* Ac = &As[cur][0];
    bf16x16 a0 = lds_a_frag(Ac, waveM * 32 + l16, half);
    bf16x16 a1 = lds_a_frag(Ac, waveM * 32 + 16 + l16, half);
#pragma unroll
    for (int nt = 0; nt < 4; ++nt) {
      int col = waveN * 64 + nt * 16 + l16;
      bf16x16 w0 = load_b_frag(Wt0, K, col, kk, half);
      bf16x16 w1 = load_b_frag(Wt1, K, col, kk, half);
      acc0[0][nt] = WMMA_BF16(a0, w0, acc0[0][nt]);
      acc0[1][nt] = WMMA_BF16(a1, w0, acc0[1][nt]);
      acc1[0][nt] = WMMA_BF16(a0, w1, acc1[0][nt]);
      acc1[1][nt] = WMMA_BF16(a1, w1, acc1[1][nt]);
    }
    cur ^= 1;
  }
#pragma unroll
  for (int mt = 0; mt < 2; ++mt) {
    int rbase = blockM + waveM * 32 + mt * 16 + half * 8;
#pragma unroll
    for (int nt = 0; nt < 4; ++nt) {
      int col = waveN * 64 + nt * 16 + l16;
      float bv0 = b0[col], bv1 = b1[col];
#pragma unroll
      for (int r = 0; r < 8; ++r) {
        int row = rbase + r;
        if (row < M) {
          float y0 = acc0[mt][nt][r] + bv0;
          float y1 = acc1[mt][nt][r] + bv1;
          if (relu) { y0 = fmaxf(y0, 0.0f); y1 = fmaxf(y1, 0.0f); }
          float o = mask[row] ? (ZMIX * y1 + (1.0f - ZMIX) * y0)
                              : (ZMIX * y0 + (1.0f - ZMIX) * y1);
          Cout[(long long)row * GH + col] = o;
        }
      }
    }
  }
}

// ---------------------------------------------------------------------------
// Small utility / GAT / norm / head kernels
// ---------------------------------------------------------------------------
__global__ void k_convert_wt(const float* __restrict__ W, __bf16* __restrict__ Wt, int K) {
  int k = blockIdx.x;          // 0..K-1
  int n = threadIdx.x;         // 0..255
  Wt[(long long)n * K + k] = f2bf(W[(long long)k * GH + n]);
}

__global__ void k_fill(float* p, long long n, float v) {
  long long i = (long long)blockIdx.x * blockDim.x + threadIdx.x;
  if (i < n) p[i] = v;
}

__global__ void k_row_bias_fill(float* __restrict__ X, const float* __restrict__ b) {
  X[(long long)blockIdx.x * GH + threadIdx.x] = b[threadIdx.x];
}

// q[c] = relu(sum_k qe[k]*W[k,c] + b[c]); one block of 256 threads
__global__ void k_gemv_q(const float* __restrict__ qe, const float* __restrict__ W,
                         const float* __restrict__ b, float* __restrict__ qv) {
  int c = threadIdx.x;
  float acc = b[c];
  for (int k = 0; k < GD; ++k) acc = fmaf(qe[k], W[(long long)k * GH + c], acc);
  qv[c] = fmaxf(acc, 0.0f);
}

// a_src[row] = h[row].att_src ; a_dst[row] = h[row].att_dst (wave per row)
__global__ void k_dots(const float* __restrict__ Hm, const float* __restrict__ asv,
                       const float* __restrict__ adv,
                       float* __restrict__ a_src, float* __restrict__ a_dst, int M) {
  int wave = threadIdx.x >> 5, lane = threadIdx.x & 31;
  int row = blockIdx.x * 8 + wave;
  if (row >= M) return;
  const float4* ph = (const float4*)(Hm + (long long)row * GH + lane * 8);
  const float4* ps = (const float4*)(asv + lane * 8);
  const float4* pd = (const float4*)(adv + lane * 8);
  float4 h0 = ph[0], h1 = ph[1], s0 = ps[0], s1 = ps[1], d0 = pd[0], d1 = pd[1];
  float ss = h0.x*s0.x + h0.y*s0.y + h0.z*s0.z + h0.w*s0.w
           + h1.x*s1.x + h1.y*s1.y + h1.z*s1.z + h1.w*s1.w;
  float sd = h0.x*d0.x + h0.y*d0.y + h0.z*d0.z + h0.w*d0.w
           + h1.x*d1.x + h1.y*d1.y + h1.z*d1.z + h1.w*d1.w;
  for (int o = 16; o > 0; o >>= 1) { ss += __shfl_down(ss, o); sd += __shfl_down(sd, o); }
  if (lane == 0) { a_src[row] = ss; a_dst[row] = sd; }
}

__global__ void k_edge_alpha_max(const int* __restrict__ ei,
                                 const float* __restrict__ a_src,
                                 const float* __restrict__ a_dst,
                                 float* __restrict__ alpha, float* __restrict__ amax) {
  int t = blockIdx.x * 256 + threadIdx.x;
  if (t >= GT) return;
  int s, d;
  if (t < GE) { s = ei[t]; d = ei[GE + t]; } else { s = d = t - GE; }
  float v = a_src[s] + a_dst[d];
  v = (v > 0.0f) ? v : SLOPE * v;
  alpha[t] = v;
  atomicMaxF(&amax[d], v);
}

__global__ void k_edge_exp_sum(const int* __restrict__ ei,
                               const float* __restrict__ alpha,
                               const float* __restrict__ amax,
                               float* __restrict__ exv, float* __restrict__ den) {
  int t = blockIdx.x * 256 + threadIdx.x;
  if (t >= GT) return;
  int d = (t < GE) ? ei[GE + t] : (t - GE);
  float e = expf(alpha[t] - amax[d]);
  exv[t] = e;
  atomicAdd(&den[d], e);
}

// out[dst] += h[src] * (ex/den[dst]); one wave per edge, 8 floats per lane
__global__ __launch_bounds__(256)
void k_edge_scatter(const int* __restrict__ ei,
                    const float* __restrict__ exv, const float* __restrict__ den,
                    const float* __restrict__ Hm, float* __restrict__ Out) {
  int wave = threadIdx.x >> 5, lane = threadIdx.x & 31;
  int t = blockIdx.x * 8 + wave;
  if (t >= GT) return;
  int s, d;
  if (t < GE) { s = ei[t]; d = ei[GE + t]; } else { s = d = t - GE; }
  float wgt = exv[t] / den[d];
  const float4* ph = (const float4*)(Hm + (long long)s * GH + lane * 8);
  float4 h0 = ph[0], h1 = ph[1];
  float* po = Out + (long long)d * GH + lane * 8;
  atomicAdd(po + 0, h0.x * wgt); atomicAdd(po + 1, h0.y * wgt);
  atomicAdd(po + 2, h0.z * wgt); atomicAdd(po + 3, h0.w * wgt);
  atomicAdd(po + 4, h1.x * wgt); atomicAdd(po + 5, h1.y * wgt);
  atomicAdd(po + 6, h1.z * wgt); atomicAdd(po + 7, h1.w * wgt);
}

// column sum & sumsq, 128 rows per block, thread = column (coalesced)
__global__ void k_colstats(const float* __restrict__ X, float* __restrict__ stats, int M) {
  int c = threadIdx.x;
  int r0 = blockIdx.x * 128;
  int rend = r0 + 128; if (rend > M) rend = M;
  float s1 = 0.0f, s2 = 0.0f;
  for (int r = r0; r < rend; ++r) {
    float v = X[(long long)r * GH + c];
    s1 += v; s2 += v * v;
  }
  atomicAdd(&stats[c], s1);
  atomicAdd(&stats[GH + c], s2);
}

// y = w*(x - mean*ms)*rsqrt(var+eps)+b ; var = E[x^2]-2*mo*mean+mo^2
__global__ void k_gnorm_apply(float* __restrict__ X, const float* __restrict__ stats,
                              const float* __restrict__ w, const float* __restrict__ b,
                              const float* __restrict__ ms, int M) {
  int c = threadIdx.x;
  long long idx = (long long)blockIdx.x * GH + c;
  float invM = 1.0f / (float)M;
  float mean = stats[c] * invM;
  float mo   = mean * ms[c];
  float var  = stats[GH + c] * invM - 2.0f * mo * mean + mo * mo;
  float x = X[idx];
  X[idx] = w[c] * (x - mo) * rsqrtf(var + EPSGN) + b[c];
}

// out[row] = dot(X[row,:256], wv) + b[0]   (wave per row)
__global__ void k_rowdot(const float* __restrict__ X, const float* __restrict__ wv,
                         const float* __restrict__ bscalar, float* __restrict__ out, int M) {
  int wave = threadIdx.x >> 5, lane = threadIdx.x & 31;
  int row = blockIdx.x * 8 + wave;
  if (row >= M) return;
  const float4* px = (const float4*)(X + (long long)row * GH + lane * 8);
  const float4* pw = (const float4*)(wv + lane * 8);
  float4 x0 = px[0], x1 = px[1], w0 = pw[0], w1 = pw[1];
  float s = x0.x*w0.x + x0.y*w0.y + x0.z*w0.z + x0.w*w0.w
          + x1.x*w1.x + x1.y*w1.y + x1.z*w1.z + x1.w*w1.w;
  for (int o = 16; o > 0; o >>= 1) s += __shfl_down(s, o);
  if (lane == 0) out[row] = s + bscalar[0];
}

__global__ void k_redmax(const float* __restrict__ x, float* gmax, int n) {
  int i = blockIdx.x * 256 + threadIdx.x;
  float v = (i < n) ? x[i] : -INFINITY;
  for (int o = 16; o > 0; o >>= 1) v = fmaxf(v, __shfl_down(v, o));
  if ((threadIdx.x & 31) == 0) atomicMaxF(gmax, v);
}

__global__ void k_expsum(const float* __restrict__ logits, const float* __restrict__ gmax,
                         float* __restrict__ probs, float* gsum, int n) {
  int i = blockIdx.x * 256 + threadIdx.x;
  float m = gmax[0];
  float e = 0.0f;
  if (i < n) { e = expf(logits[i] - m); probs[i] = e; }
  for (int o = 16; o > 0; o >>= 1) e += __shfl_down(e, o);
  if ((threadIdx.x & 31) == 0) atomicAdd(gsum, e);
}

__global__ void k_norm_entropy(float* __restrict__ probs, const float* __restrict__ gsum,
                               float* ent, int n) {
  int i = blockIdx.x * 256 + threadIdx.x;
  float inv = 1.0f / gsum[0];
  float h = 0.0f;
  if (i < n) {
    float p = probs[i] * inv;
    probs[i] = p;
    h = -p * logf(p + 1e-10f);
  }
  for (int o = 16; o > 0; o >>= 1) h += __shfl_down(h, o);
  if ((threadIdx.x & 31) == 0) atomicAdd(ent, h);
}

__global__ void k_meanadd(const float* __restrict__ v, float* out, int n) {
  int i = blockIdx.x * 256 + threadIdx.x;
  float s = (i < n) ? v[i] : 0.0f;
  for (int o = 16; o > 0; o >>= 1) s += __shfl_down(s, o);
  if ((threadIdx.x & 31) == 0) atomicAdd(out, s * (1.0f / (float)n));
}

// ============================================================================
// Host side
// ============================================================================
extern "C" void kernel_launch(void* const* d_in, const int* in_sizes, int n_in,
                              void* d_out_v, int out_size, void* d_ws, size_t ws_size,
                              hipStream_t stream) {
  (void)in_sizes; (void)n_in; (void)out_size; (void)ws_size;
  auto F = [&](int i) -> const float* { return (const float*)d_in[i]; };

  // --- params (sorted-pytree flatten; see header comment) ---
  const float* LW[2][23];
  for (int l = 0; l < 2; ++l)
    for (int j = 0; j < 23; ++j) LW[l][j] = F(2 + 23 * l + j);
  const float* node_b = F(48); const float* node_w = F(49);
  const float* ph_b   = F(50); const float* ph_w   = F(51);
  const float* q_b    = F(52); const float* q_w    = F(53);
  const float* v1_b   = F(54); const float* v1_w   = F(55);
  const float* v2_b   = F(56); const float* v2_w   = F(57);
  const float* x_in   = F(58);
  const int*   ei     = (const int*)d_in[59];
  const float* qe     = F(61);
  const unsigned char* mask = (const unsigned char*)d_in[62];

  // --- workspace carve-out ---
  char* wsb = (char*)d_ws;
  size_t off = 0;
  auto alloc = [&](size_t bytes) -> void* {
    off = (off + 255) & ~(size_t)255;
    void* p = wsb + off;
    off += bytes;
    return p;
  };
  float* bufA  = (float*)alloc((size_t)GN * GH * 4);   // x
  float* bufB  = (float*)alloc((size_t)GN * GH * 4);   // x_comb (x_in of conv)
  float* bufC  = (float*)alloc((size_t)GN * GH * 4);   // xm -> gat out -> normed
  float* bufD  = (float*)alloc((size_t)GN * GH * 4);   // h
  float* qvec  = (float*)alloc(GH * 4);
  float* asrc  = (float*)alloc(GN * 4);
  float* adst  = (float*)alloc(GN * 4);
  float* amax  = (float*)alloc(GN * 4);
  float* den   = (float*)alloc(GN * 4);
  float* alpha = (float*)alloc((size_t)GT * 4);
  float* exv   = (float*)alloc((size_t)GT * 4);
  float* stats = (float*)alloc(2 * GH * 4);
  float* lvec  = (float*)alloc(GN * 4);                // logits, then value vec
  float* gmax  = (float*)alloc(4);
  float* gsum  = (float*)alloc(4);
  __bf16* wt_node = (__bf16*)alloc((size_t)GD * GH * 2);
  __bf16* wt_nq[2], *wt_t0[2], *wt_t1[2], *wt_gat[2], *wt_c0[2], *wt_c1[2];
  for (int l = 0; l < 2; ++l) {
    wt_nq[l]  = (__bf16*)alloc((size_t)2 * GH * GH * 2);
    wt_t0[l]  = (__bf16*)alloc((size_t)GH * GH * 2);
    wt_t1[l]  = (__bf16*)alloc((size_t)GH * GH * 2);
    wt_gat[l] = (__bf16*)alloc((size_t)GH * GH * 2);
    wt_c0[l]  = (__bf16*)alloc((size_t)2 * GH * GH * 2);
    wt_c1[l]  = (__bf16*)alloc((size_t)2 * GH * GH * 2);
  }
  __bf16* wt_v1 = (__bf16*)alloc((size_t)GH * GH * 2);

  float* out   = (float*)d_out_v;
  float* probs = out;
  float* sval  = out + GN;
  float* xout  = out + GN + 1;
  float* ent   = out + GN + 1 + (size_t)GN * GH;

  const int MG   = (GN + 63) / 64;     // GEMM grid
  const int RG   = (GN + 7) / 8;       // wave-per-row grid
  const int EG   = (GT + 255) / 256;   // edge grid
  const int SG   = (GT + 7) / 8;       // scatter grid
  const int NG   = (GN + 255) / 256;   // 1-thread-per-node grid
  const int CSG  = (GN + 127) / 128;   // colstats grid

  // --- weight conversion (f32 [K,256] -> bf16 transposed [256,K]) ---
  k_convert_wt<<<GD, 256, 0, stream>>>(node_w, wt_node, GD);
  for (int l = 0; l < 2; ++l) {
    k_convert_wt<<<2 * GH, 256, 0, stream>>>(LW[l][18], wt_nq[l],  2 * GH);
    k_convert_wt<<<GH,     256, 0, stream>>>(LW[l][20], wt_t0[l],  GH);
    k_convert_wt<<<GH,     256, 0, stream>>>(LW[l][22], wt_t1[l],  GH);
    k_convert_wt<<<GH,     256, 0, stream>>>(LW[l][10], wt_gat[l], GH);
    k_convert_wt<<<2 * GH, 256, 0, stream>>>(LW[l][3],  wt_c0[l],  2 * GH);
    k_convert_wt<<<2 * GH, 256, 0, stream>>>(LW[l][5],  wt_c1[l],  2 * GH);
  }
  k_convert_wt<<<GH, 256, 0, stream>>>(v1_w, wt_v1, GH);

  // --- q = relu(question_input(qe)) ---
  k_gemv_q<<<1, 256, 0, stream>>>(qe, q_w, q_b, qvec);

  // --- x = relu(node_input(x_))  (edge_input path is dead -> skipped) ---
  k_gemm<<<MG, 256, 0, stream>>>(x_in, GD, nullptr, 0, GD,
                                 wt_node, GD, node_b, bufA, GN, 1);

  for (int l = 0; l < 2; ++l) {
    // x_comb = relu(nq_mix([x | q]))
    k_gemm<<<MG, 256, 0, stream>>>(bufA, GH, qvec, 0, GH,
                                   wt_nq[l], 2 * GH, LW[l][17], bufB, GN, 1);
    // xm = mix(mask, relu(trans0(x_comb)), relu(trans1(x_comb)))
    k_gemm_dual_mix<<<MG, 256, 0, stream>>>(bufB, GH, nullptr, 0, GH,
                                            wt_t0[l], wt_t1[l], GH,
                                            LW[l][19], LW[l][21], mask, bufC, GN, 1);
    // h = gat_lin(xm)
    k_gemm<<<MG, 256, 0, stream>>>(bufC, GH, nullptr, 0, GH,
                                   wt_gat[l], GH, LW[l][9], bufD, GN, 0);
    // attention scalars
    k_dots<<<RG, 256, 0, stream>>>(bufD, LW[l][1], LW[l][0], asrc, adst, GN);
    // GAT softmax-aggregate
    k_fill<<<NG, 256, 0, stream>>>(amax, GN, -INFINITY);
    k_fill<<<NG, 256, 0, stream>>>(den, GN, 0.0f);
    k_edge_alpha_max<<<EG, 256, 0, stream>>>(ei, asrc, adst, alpha, amax);
    k_edge_exp_sum<<<EG, 256, 0, stream>>>(ei, alpha, amax, exv, den);
    k_row_bias_fill<<<GN, 256, 0, stream>>>(bufC, LW[l][8]);   // out = gat_bias
    k_edge_scatter<<<SG, 256, 0, stream>>>(ei, exv, den, bufD, bufC);
    // graph_norm gn_in (in place on bufC)
    k_fill<<<2, 256, 0, stream>>>(stats, 2 * GH, 0.0f);
    k_colstats<<<CSG, 256, 0, stream>>>(bufC, stats, GN);
    k_gnorm_apply<<<GN, 256, 0, stream>>>(bufC, stats, LW[l][13], LW[l][11], LW[l][12], GN);
    // x_new = mix(mask, comb0([x|x_in]), comb1([x|x_in]))  -> bufA
    k_gemm_dual_mix<<<MG, 256, 0, stream>>>(bufC, GH, bufB, GH, GH,
                                            wt_c0[l], wt_c1[l], 2 * GH,
                                            LW[l][2], LW[l][4], mask, bufA, GN, 0);
    // graph_norm gn_out (in place on bufA)
    k_fill<<<2, 256, 0, stream>>>(stats, 2 * GH, 0.0f);
    k_colstats<<<CSG, 256, 0, stream>>>(bufA, stats, GN);
    k_gnorm_apply<<<GN, 256, 0, stream>>>(bufA, stats, LW[l][16], LW[l][14], LW[l][15], GN);
  }

  // --- policy head: softmax + entropy ---
  k_rowdot<<<RG, 256, 0, stream>>>(bufA, ph_w, ph_b, lvec, GN);
  k_fill<<<1, 256, 0, stream>>>(gmax, 1, -INFINITY);
  k_fill<<<1, 256, 0, stream>>>(gsum, 1, 0.0f);
  k_fill<<<1, 256, 0, stream>>>(sval, 1, 0.0f);
  k_fill<<<1, 256, 0, stream>>>(ent, 1, 0.0f);
  k_redmax<<<NG, 256, 0, stream>>>(lvec, gmax, GN);
  k_expsum<<<NG, 256, 0, stream>>>(lvec, gmax, probs, gsum, GN);
  k_norm_entropy<<<NG, 256, 0, stream>>>(probs, gsum, ent, GN);

  // --- x output ---
  hipMemcpyAsync(xout, bufA, (size_t)GN * GH * 4, hipMemcpyDeviceToDevice, stream);

  // --- value head: state_value = mean(value2(relu(value1(x)))) ---
  k_gemm<<<MG, 256, 0, stream>>>(bufA, GH, nullptr, 0, GH,
                                 wt_v1, GH, v1_b, bufB, GN, 1);
  k_rowdot<<<RG, 256, 0, stream>>>(bufB, v2_w, v2_b, lvec, GN);
  k_meanadd<<<NG, 256, 0, stream>>>(lvec, sval, GN);
}
